// SummarizationTransformer_31988916421106
// MI455X (gfx1250) — compile-verified, matrix operand order checked by reading, not run on hardware
//
#include <hip/hip_runtime.h>
#include <math.h>

// ---------------------------------------------------------------------------
// CDNA5 (gfx1250) wave32 WMMA transformer forward pass.
// All GEMMs use v_wmma_f32_16x16x32_bf16 (fp32 accumulate, bf16 operands
// converted inline with v_cvt_pk_bf16_f32). Attention is fused flash-style
// (never materializes the [B,H,Sq,Sk] score tensor).
// ---------------------------------------------------------------------------

typedef __bf16 bf16_t;
typedef bf16_t v16bf __attribute__((ext_vector_type(16)));
typedef bf16_t v2bf  __attribute__((ext_vector_type(2)));
typedef float  v8f   __attribute__((ext_vector_type(8)));

union FragBF { v16bf v; unsigned int u[8]; };

__device__ __forceinline__ unsigned short f2bf(float f) {
  return __builtin_bit_cast(unsigned short, (__bf16)f);   // native cvt, RTNE
}
// Packed pair -> single v_cvt_pk_bf16_f32 (no bit-fixup churn).
__device__ __forceinline__ unsigned int pack2bf(float lo, float hi) {
  v2bf p; p.x = (__bf16)lo; p.y = (__bf16)hi;
  return __builtin_bit_cast(unsigned int, p);
}
// Element-pair offset inside a 32-wide K chunk for 16-bit A/B fragments
// (ISA 7.12.2: VGPR j<4 -> K = half*8 + 2j, j>=4 -> K = 16 + half*8 + 2(j-4)).
__device__ __forceinline__ int kpair(int j, int half) {
  return (j < 4) ? (half * 8 + 2 * j) : (16 + half * 8 + 2 * (j - 4));
}
__device__ __forceinline__ v8f wmma_bf16(const FragBF& a, const FragBF& b, v8f c) {
  return __builtin_amdgcn_wmma_f32_16x16x32_bf16(false, a.v, false, b.v,
                                                 (short)0, c, false, false);
}
// Convert a float4 into two packed bf16 pairs (2x v_cvt_pk_bf16_f32).
__device__ __forceinline__ uint2 cvt4(float4 f) {
  uint2 r; r.x = pack2bf(f.x, f.y); r.y = pack2bf(f.z, f.w); return r;
}

// ---------------------------------------------------------------------------
// GEMM: C[M,N] = act(A[M,K] (fp32) x W[K,N] (fp32, converted to bf16) + bias)
// Block tile 64x128, 8 waves (wave32), each wave computes 16x64.
// ---------------------------------------------------------------------------
__global__ __launch_bounds__(256) void k_gemm(
    const float* __restrict__ A, const float* __restrict__ W,
    const float* __restrict__ bias, float* __restrict__ C,
    int M, int N, int K, int relu)
{
  __shared__ unsigned short As[64][32];    // [m][k]  bf16
  __shared__ unsigned short Bt[128][32];   // [n][k]  bf16 (transposed stage)

  const int tid  = threadIdx.x;
  const int w    = tid >> 5, lane = tid & 31;
  const int half = lane >> 4, l16 = lane & 15;
  const int wm   = w & 3,  wn   = w >> 2;          // 4 x 2 wave grid
  const int m0   = blockIdx.y * 64, n0 = blockIdx.x * 128;

  // Per-thread staging coordinates (loop-invariant)
  const int amm[2] = { (tid + 0)   >> 3, (tid + 256) >> 3 };
  const int aq4[2] = { (tid + 0)   &  7, (tid + 256) &  7 };
  int wkk[4], wn4[4];
  #pragma unroll
  for (int i = 0; i < 4; ++i) { int e = tid + i * 256; wkk[i] = e >> 5; wn4[i] = e & 31; }

  v8f acc[4] = {};

  for (int k0 = 0; k0 < K; k0 += 32) {
    __syncthreads();
    // ---- Load phase: issue all b128 loads back-to-back (6 in flight) ----
    float4 fa[2], fw[4];
    #pragma unroll
    for (int i = 0; i < 2; ++i)
      fa[i] = *(const float4*)&A[(size_t)(m0 + amm[i]) * K + k0 + aq4[i] * 4];
    #pragma unroll
    for (int i = 0; i < 4; ++i)
      fw[i] = *(const float4*)&W[(size_t)(k0 + wkk[i]) * N + n0 + wn4[i] * 4];
    // ---- Convert + store phase ----
    #pragma unroll
    for (int i = 0; i < 2; ++i)
      *(uint2*)&As[amm[i]][aq4[i] * 4] = cvt4(fa[i]);
    #pragma unroll
    for (int i = 0; i < 4; ++i) {
      Bt[wn4[i] * 4 + 0][wkk[i]] = f2bf(fw[i].x);
      Bt[wn4[i] * 4 + 1][wkk[i]] = f2bf(fw[i].y);
      Bt[wn4[i] * 4 + 2][wkk[i]] = f2bf(fw[i].z);
      Bt[wn4[i] * 4 + 3][wkk[i]] = f2bf(fw[i].w);
    }
    if (k0 + 32 < K)
      __builtin_prefetch(&W[(size_t)(k0 + 32) * N + n0 + (tid & 127)], 0, 1);
    __syncthreads();

    FragBF a;
    const int mrow = wm * 16 + l16;
    #pragma unroll
    for (int j = 0; j < 8; ++j)
      a.u[j] = *(const unsigned int*)&As[mrow][kpair(j, half)];

    const int ncb = wn * 64 + l16;
    #pragma unroll
    for (int t = 0; t < 4; ++t) {
      FragBF b;
      #pragma unroll
      for (int j = 0; j < 8; ++j)
        b.u[j] = *(const unsigned int*)&Bt[ncb + t * 16][kpair(j, half)];
      acc[t] = wmma_bf16(a, b, acc[t]);
    }
  }

  // Epilogue: C layout — lane col = lane%16, rows = r + 8*(lane/16)
  const int ccol = n0 + wn * 64 + l16;
  #pragma unroll
  for (int t = 0; t < 4; ++t) {
    #pragma unroll
    for (int r = 0; r < 8; ++r) {
      int row = m0 + wm * 16 + r + 8 * half;
      float v = acc[t][r] + bias[ccol + t * 16];
      if (relu) v = fmaxf(v, 0.f);
      C[(size_t)row * N + ccol + t * 16] = v;
    }
  }
}

// ---------------------------------------------------------------------------
// Fused multi-head attention (flash style), dk = 64, heads packed in D.
// grid = (Sq/64, H, B), block = 128 (4 waves x 16 query rows).
// ---------------------------------------------------------------------------
__global__ __launch_bounds__(128) void k_attn(
    const float* __restrict__ Qg, const float* __restrict__ Kg,
    const float* __restrict__ Vg, float* __restrict__ Og,
    int Sq, int Sk, int D, int causal)
{
  __shared__ unsigned short Ks[32][64];     // [key][dk]
  __shared__ unsigned short Vs[64][32];     // [dk][key] (transposed)
  __shared__ unsigned short Ps[4][16][32];  // per-wave P tile [q][key]

  const int tid  = threadIdx.x;
  const int w    = tid >> 5, lane = tid & 31;
  const int half = lane >> 4, l16 = lane & 15;
  const int h = blockIdx.y, b = blockIdx.z;
  const int qbase = blockIdx.x * 64 + w * 16;

  const float* Qp = Qg + ((size_t)b * Sq) * D + h * 64;
  const float* Kp = Kg + ((size_t)b * Sk) * D + h * 64;
  const float* Vp = Vg + ((size_t)b * Sk) * D + h * 64;
  float*       Op = Og + ((size_t)b * Sq) * D + h * 64;

  // Q fragments (A layout): lane row = qbase + lane%16, two dk chunks of 32.
  // Pairs j=0..3 and j=4..7 each cover 8 contiguous K values -> float4 loads.
  FragBF qa[2];
  {
    const size_t qoff = (size_t)(qbase + l16) * D;
    float4 qf[8];
    #pragma unroll
    for (int c = 0; c < 2; ++c)
      #pragma unroll
      for (int g = 0; g < 2; ++g) {
        int base = c * 32 + g * 16 + half * 8;
        qf[(c * 2 + g) * 2 + 0] = *(const float4*)&Qp[qoff + base];
        qf[(c * 2 + g) * 2 + 1] = *(const float4*)&Qp[qoff + base + 4];
      }
    #pragma unroll
    for (int c = 0; c < 2; ++c)
      #pragma unroll
      for (int g = 0; g < 2; ++g) {
        uint2 a0 = cvt4(qf[(c * 2 + g) * 2 + 0]);
        uint2 a1 = cvt4(qf[(c * 2 + g) * 2 + 1]);
        qa[c].u[g * 4 + 0] = a0.x; qa[c].u[g * 4 + 1] = a0.y;
        qa[c].u[g * 4 + 2] = a1.x; qa[c].u[g * 4 + 3] = a1.y;
      }
  }

  // Per-thread staging coordinates (loop-invariant)
  int skey[4], sq4[4];
  #pragma unroll
  for (int i = 0; i < 4; ++i) { int e = tid + i * 128; skey[i] = e >> 4; sq4[i] = e & 15; }

  float mrun[8], lrun[8];
  v8f oacc[4] = {};
  #pragma unroll
  for (int r = 0; r < 8; ++r) { mrun[r] = -1e30f; lrun[r] = 0.f; }

  const int ktiles = causal ? ((blockIdx.x * 64 + 64) / 32) : (Sk / 32);
  for (int kt = 0; kt < ktiles; ++kt) {
    const int kb = kt * 32;
    __syncthreads();
    // ---- Load phase: 8 b128 loads in flight, then convert/store ----
    float4 kf[4], vf[4];
    #pragma unroll
    for (int i = 0; i < 4; ++i) {
      size_t goff = (size_t)(kb + skey[i]) * D + sq4[i] * 4;
      kf[i] = *(const float4*)&Kp[goff];
      vf[i] = *(const float4*)&Vp[goff];
    }
    #pragma unroll
    for (int i = 0; i < 4; ++i) {
      *(uint2*)&Ks[skey[i]][sq4[i] * 4] = cvt4(kf[i]);
      Vs[sq4[i] * 4 + 0][skey[i]] = f2bf(vf[i].x);
      Vs[sq4[i] * 4 + 1][skey[i]] = f2bf(vf[i].y);
      Vs[sq4[i] * 4 + 2][skey[i]] = f2bf(vf[i].z);
      Vs[sq4[i] * 4 + 3][skey[i]] = f2bf(vf[i].w);
    }
    __syncthreads();

    // Scores: 16q x 32k = two 16x16 WMMA accumulators over 2 dk chunks
    v8f s0 = {}, s1 = {};
    #pragma unroll
    for (int c = 0; c < 2; ++c) {
      FragBF kb0, kb1;
      #pragma unroll
      for (int j = 0; j < 8; ++j) {
        int kp = c * 32 + kpair(j, half);
        kb0.u[j] = *(const unsigned int*)&Ks[l16][kp];
        kb1.u[j] = *(const unsigned int*)&Ks[16 + l16][kp];
      }
      s0 = wmma_bf16(qa[c], kb0, s0);
      s1 = wmma_bf16(qa[c], kb1, s1);
    }

    // Online softmax (row = r + 8*half, cols spread over 16-lane half-groups)
    #pragma unroll
    for (int r = 0; r < 8; ++r) {
      float v0 = s0[r] * 0.125f;   // 1/sqrt(64)
      float v1 = s1[r] * 0.125f;
      if (causal) {
        int qr = qbase + r + 8 * half;
        if (kb + l16 > qr)      v0 = -1e30f;
        if (kb + 16 + l16 > qr) v1 = -1e30f;
      }
      float mt = fmaxf(v0, v1);
      #pragma unroll
      for (int o = 1; o < 16; o <<= 1) mt = fmaxf(mt, __shfl_xor(mt, o, 32));
      float mn = fmaxf(mrun[r], mt);
      float p0 = __expf(v0 - mn), p1 = __expf(v1 - mn);
      float ts = p0 + p1;
      #pragma unroll
      for (int o = 1; o < 16; o <<= 1) ts += __shfl_xor(ts, o, 32);
      float alpha = __expf(mrun[r] - mn);
      lrun[r] = lrun[r] * alpha + ts;
      mrun[r] = mn;
      #pragma unroll
      for (int cc = 0; cc < 4; ++cc) oacc[cc][r] *= alpha;
      Ps[w][r + 8 * half][l16]      = f2bf(p0);
      Ps[w][r + 8 * half][16 + l16] = f2bf(p1);
    }
    // Wave-local LDS RAW fence (C-layout -> A-layout transpose round-trip)
    asm volatile("s_wait_dscnt 0x0" ::: "memory");

    // P as A fragment, V as B fragment (4 dk column subtiles)
    FragBF pa;
    #pragma unroll
    for (int j = 0; j < 8; ++j)
      pa.u[j] = *(const unsigned int*)&Ps[w][l16][kpair(j, half)];
    #pragma unroll
    for (int cc = 0; cc < 4; ++cc) {
      FragBF vb;
      #pragma unroll
      for (int j = 0; j < 8; ++j)
        vb.u[j] = *(const unsigned int*)&Vs[cc * 16 + l16][kpair(j, half)];
      oacc[cc] = wmma_bf16(pa, vb, oacc[cc]);
    }
  }

  #pragma unroll
  for (int cc = 0; cc < 4; ++cc)
    #pragma unroll
    for (int r = 0; r < 8; ++r) {
      int qr = qbase + r + 8 * half;
      Op[(size_t)qr * D + cc * 16 + l16] = oacc[cc][r] / lrun[r];
    }
}

// ---------------------------------------------------------------------------
// Residual add + LayerNorm (one block per token row, D = 1024)
// ---------------------------------------------------------------------------
__global__ __launch_bounds__(256) void k_add_ln(
    const float* __restrict__ X, const float* __restrict__ S,
    const float* __restrict__ g, const float* __restrict__ be,
    float* __restrict__ Out, int D)
{
  const int row = blockIdx.x, tid = threadIdx.x;
  __shared__ float red[256];
  float v[4];
  float s = 0.f;
  #pragma unroll
  for (int i = 0; i < 4; ++i) {
    int d = tid + i * 256;
    v[i] = X[(size_t)row * D + d] + S[(size_t)row * D + d];
    s += v[i];
  }
  red[tid] = s; __syncthreads();
  for (int o = 128; o > 0; o >>= 1) { if (tid < o) red[tid] += red[tid + o]; __syncthreads(); }
  float mean = red[0] / (float)D;
  __syncthreads();
  float s2 = 0.f;
  #pragma unroll
  for (int i = 0; i < 4; ++i) { float dlt = v[i] - mean; s2 += dlt * dlt; }
  red[tid] = s2; __syncthreads();
  for (int o = 128; o > 0; o >>= 1) { if (tid < o) red[tid] += red[tid + o]; __syncthreads(); }
  float inv = rsqrtf(red[0] / (float)D + 1e-5f);
  #pragma unroll
  for (int i = 0; i < 4; ++i) {
    int d = tid + i * 256;
    Out[(size_t)row * D + d] = (v[i] - mean) * inv * g[d] + be[d];
  }
}

// ---------------------------------------------------------------------------
// Embedding gather * sqrt(D) + sinusoidal positional encoding
// ---------------------------------------------------------------------------
__global__ __launch_bounds__(256) void k_embed(
    const int* __restrict__ ids, const float* __restrict__ emb,
    float* __restrict__ out, int S, int D, float scale)
{
  const int t = blockIdx.x;
  const int pos = t % S;
  const int id = ids[t];
  const float c = -9.2103403719761836f / (float)D;  // -ln(10000)/D
  #pragma unroll
  for (int i = 0; i < 4; ++i) {
    int d = threadIdx.x + i * 256;
    float ang = (float)pos * __expf(c * (float)(d & ~1));
    float pe = (d & 1) ? __cosf(ang) : __sinf(ang);
    out[(size_t)t * D + d] = emb[(size_t)id * D + d] * scale + pe;
  }
}

// ---------------------------------------------------------------------------
// Host orchestration
// ---------------------------------------------------------------------------
extern "C" void kernel_launch(void* const* d_in, const int* in_sizes, int n_in,
                              void* d_out, int out_size, void* d_ws, size_t ws_size,
                              hipStream_t stream)
{
  (void)in_sizes; (void)n_in; (void)out_size; (void)ws_size;
  const int Bb = 4, Ssrc = 1024, Stgt = 256, D = 1024, F = 4096, Lh = 6, Hn = 16, Vv = 32000;
  const int Tsrc = Bb * Ssrc;   // 4096 token rows
  const int Ttgt = Bb * Stgt;   // 1024 token rows

  const int*   input_ids  = (const int*)d_in[0];
  const int*   target_ids = (const int*)d_in[1];
  const float* emb_W      = (const float*)d_in[2];
  const float* enc_attn_w = (const float*)d_in[3];
  const float* enc_attn_b = (const float*)d_in[4];
  const float* enc_ff_w1  = (const float*)d_in[5];
  const float* enc_ff_b1  = (const float*)d_in[6];
  const float* enc_ff_w2  = (const float*)d_in[7];
  const float* enc_ff_b2  = (const float*)d_in[8];
  const float* enc_ln_g   = (const float*)d_in[9];
  const float* enc_ln_b   = (const float*)d_in[10];
  const float* dec_self_w = (const float*)d_in[11];
  const float* dec_self_b = (const float*)d_in[12];
  const float* dec_cross_w= (const float*)d_in[13];
  const float* dec_cross_b= (const float*)d_in[14];
  const float* dec_ff_w1  = (const float*)d_in[15];
  const float* dec_ff_b1  = (const float*)d_in[16];
  const float* dec_ff_w2  = (const float*)d_in[17];
  const float* dec_ff_b2  = (const float*)d_in[18];
  const float* dec_ln_g   = (const float*)d_in[19];
  const float* dec_ln_b   = (const float*)d_in[20];
  const float* out_W      = (const float*)d_in[21];
  const float* out_b      = (const float*)d_in[22];

  char* ws = (char*)d_ws;
  size_t off = 0;
  auto alloc = [&](size_t bytes) -> void* {
    off = (off + 255) & ~(size_t)255;
    void* p = ws + off; off += bytes; return p;
  };
  float* X   = (float*)alloc((size_t)Tsrc * D * 4);  // encoder activations
  float* Y   = (float*)alloc((size_t)Ttgt * D * 4);  // decoder activations
  float* T1  = (float*)alloc((size_t)Tsrc * D * 4);  // sublayer output
  float* Qb  = (float*)alloc((size_t)Tsrc * D * 4);
  float* Kb  = (float*)alloc((size_t)Tsrc * D * 4);
  float* Vb  = (float*)alloc((size_t)Tsrc * D * 4);
  float* CTX = (float*)alloc((size_t)Tsrc * D * 4);
  float* HID = (float*)alloc((size_t)Tsrc * F * 4);

  auto gemm = [&](const float* A, const float* W, const float* bias, float* C,
                  int M, int N, int K, int relu) {
    dim3 g(N / 128, M / 64);
    k_gemm<<<g, 256, 0, stream>>>(A, W, bias, C, M, N, K, relu);
  };
  auto addln = [&](const float* Xr, const float* Sr, const float* g,
                   const float* be, float* O, int rows) {
    k_add_ln<<<rows, 256, 0, stream>>>(Xr, Sr, g, be, O, D);
  };

  // ---------------- Encoder ----------------
  k_embed<<<Tsrc, 256, 0, stream>>>(input_ids, emb_W, X, Ssrc, D, 32.0f);
  for (int l = 0; l < Lh; ++l) {
    const float* aw = enc_attn_w + (size_t)l * 4 * D * D;
    const float* ab = enc_attn_b + (size_t)l * 4 * D;
    gemm(X, aw + 0 * (size_t)D * D, ab + 0 * D, Qb, Tsrc, D, D, 0);
    gemm(X, aw + 1 * (size_t)D * D, ab + 1 * D, Kb, Tsrc, D, D, 0);
    gemm(X, aw + 2 * (size_t)D * D, ab + 2 * D, Vb, Tsrc, D, D, 0);
    k_attn<<<dim3(Ssrc / 64, Hn, Bb), 128, 0, stream>>>(Qb, Kb, Vb, CTX, Ssrc, Ssrc, D, 0);
    gemm(CTX, aw + 3 * (size_t)D * D, ab + 3 * D, T1, Tsrc, D, D, 0);
    addln(X, T1, enc_ln_g + (size_t)(l * 2 + 0) * D, enc_ln_b + (size_t)(l * 2 + 0) * D, X, Tsrc);
    gemm(X, enc_ff_w1 + (size_t)l * D * F, enc_ff_b1 + (size_t)l * F, HID, Tsrc, F, D, 1);
    gemm(HID, enc_ff_w2 + (size_t)l * F * D, enc_ff_b2 + (size_t)l * D, T1, Tsrc, D, F, 0);
    addln(X, T1, enc_ln_g + (size_t)(l * 2 + 1) * D, enc_ln_b + (size_t)(l * 2 + 1) * D, X, Tsrc);
  }

  // ---------------- Decoder ----------------
  k_embed<<<Ttgt, 256, 0, stream>>>(target_ids, emb_W, Y, Stgt, D, 32.0f);
  for (int l = 0; l < Lh; ++l) {
    // causal self-attention
    const float* sw = dec_self_w + (size_t)l * 4 * D * D;
    const float* sb = dec_self_b + (size_t)l * 4 * D;
    gemm(Y, sw + 0 * (size_t)D * D, sb + 0 * D, Qb, Ttgt, D, D, 0);
    gemm(Y, sw + 1 * (size_t)D * D, sb + 1 * D, Kb, Ttgt, D, D, 0);
    gemm(Y, sw + 2 * (size_t)D * D, sb + 2 * D, Vb, Ttgt, D, D, 0);
    k_attn<<<dim3(Stgt / 64, Hn, Bb), 128, 0, stream>>>(Qb, Kb, Vb, CTX, Stgt, Stgt, D, 1);
    gemm(CTX, sw + 3 * (size_t)D * D, sb + 3 * D, T1, Ttgt, D, D, 0);
    addln(Y, T1, dec_ln_g + (size_t)(l * 3 + 0) * D, dec_ln_b + (size_t)(l * 3 + 0) * D, Y, Ttgt);
    // cross-attention against encoder output X
    const float* cw = dec_cross_w + (size_t)l * 4 * D * D;
    const float* cb = dec_cross_b + (size_t)l * 4 * D;
    gemm(Y, cw + 0 * (size_t)D * D, cb + 0 * D, Qb, Ttgt, D, D, 0);
    gemm(X, cw + 1 * (size_t)D * D, cb + 1 * D, Kb, Tsrc, D, D, 0);
    gemm(X, cw + 2 * (size_t)D * D, cb + 2 * D, Vb, Tsrc, D, D, 0);
    k_attn<<<dim3(Stgt / 64, Hn, Bb), 128, 0, stream>>>(Qb, Kb, Vb, CTX, Stgt, Ssrc, D, 0);
    gemm(CTX, cw + 3 * (size_t)D * D, cb + 3 * D, T1, Ttgt, D, D, 0);
    addln(Y, T1, dec_ln_g + (size_t)(l * 3 + 1) * D, dec_ln_b + (size_t)(l * 3 + 1) * D, Y, Ttgt);
    // FFN
    gemm(Y, dec_ff_w1 + (size_t)l * D * F, dec_ff_b1 + (size_t)l * F, HID, Ttgt, F, D, 1);
    gemm(HID, dec_ff_w2 + (size_t)l * F * D, dec_ff_b2 + (size_t)l * D, T1, Ttgt, D, F, 0);
    addln(Y, T1, dec_ln_g + (size_t)(l * 3 + 2) * D, dec_ln_b + (size_t)(l * 3 + 2) * D, Y, Ttgt);
  }

  // ---------------- Vocab projection ----------------
  gemm(Y, out_W, out_b, (float*)d_out, Ttgt, Vv, D, 0);
}
